// BinaryContrastiveLoss_19576460935642
// MI455X (gfx1250) — compile-verified, matrix operand order checked by reading, not run on hardware
//
#include <hip/hip_runtime.h>
#include <hip/hip_bf16.h>

// Problem constants (match reference)
#define BB 8
#define NN 4096
#define DD 128
#define PP 16
#define QQ 32

typedef __attribute__((ext_vector_type(16))) _Float16 v16h;
typedef __attribute__((ext_vector_type(8)))  _Float16 v8h_t;
typedef __attribute__((ext_vector_type(4)))  _Float16 v4h;
typedef __attribute__((ext_vector_type(8)))  float    v8f;
typedef __attribute__((ext_vector_type(4)))  float    v4f;

union V16H { v16h v; v8h_t h[2]; };

// ---------------------------------------------------------------------------
// Kernel 0: zero the per-batch accumulators (must run every call: the harness
// does not re-zero the workspace between graph replays).
// ---------------------------------------------------------------------------
__global__ void bcl_init_kernel(float* sums) {
    int t = threadIdx.x;
    if (t < 2 * BB) sums[t] = 0.0f;
}

// ---------------------------------------------------------------------------
// Kernel 1: L2-normalize each D=128 row and store as f16.
// One wave per row; lane holds 4 consecutive floats.
// ---------------------------------------------------------------------------
__global__ void __launch_bounds__(256)
bcl_normalize_kernel(const float* __restrict__ feat, _Float16* __restrict__ out) {
    int lane = threadIdx.x & 31;
    int wave = threadIdx.x >> 5;
    int row  = blockIdx.x * 8 + wave;          // < BB*NN
    size_t base = (size_t)row * DD + (size_t)lane * 4;
    v4f x = *(const v4f*)(feat + base);
    float ss = x[0]*x[0] + x[1]*x[1] + x[2]*x[2] + x[3]*x[3];
    #pragma unroll
    for (int o = 16; o > 0; o >>= 1) ss += __shfl_xor(ss, o, 32);
    float inv = 1.0f / fmaxf(sqrtf(ss), 1e-12f);
    v4h h;
    h[0] = (_Float16)(x[0] * inv);
    h[1] = (_Float16)(x[1] * inv);
    h[2] = (_Float16)(x[2] * inv);
    h[3] = (_Float16)(x[3] * inv);
    *(v4h*)(out + base) = h;
}

// ---------------------------------------------------------------------------
// Kernel 2: per-row contrastive loss via WMMA.
// One wave per row n. A = 16 gathered f16 rows (tile), B = row vector
// broadcast into all 16 columns, C/D = f32 accumulator; K accumulated over
// 4 chunks of 32. All 16 columns of D carry identical scores, so every lane
// stores its accumulator slice unconditionally (same-value LDS writes are
// benign) -> no exec-mask divergence. Then wave32 shuffle softmax/log1p
// reduction and per-batch global f32 atomics.
// ---------------------------------------------------------------------------
__global__ void __launch_bounds__(256)
bcl_main_kernel(const _Float16* __restrict__ f16,
                const int* __restrict__ posIdx,
                const int* __restrict__ negIdx,
                float* __restrict__ lossSum,
                float* __restrict__ maskSum) {
    __shared__ float scores[8][48];

    int lane = threadIdx.x & 31;
    int wave = threadIdx.x >> 5;
    int row  = blockIdx.x * 8 + wave;          // flat (b,n), < BB*NN
    int b    = row >> 12;                      // N = 4096
    int m    = lane & 15;
    int off  = (lane < 16) ? 0 : 8;            // ISA 16-bit A/B per-lane K split

    const _Float16* myRow     = f16 + (size_t)row * DD;
    const _Float16* batchBase = f16 + ((size_t)b * NN) * DD;

    // B-matrix chunks: broadcast my row vector into all 16 columns.
    // Only the per-lane K distribution matters since all columns are equal.
    V16H Bm[4];
    #pragma unroll
    for (int c = 0; c < 4; ++c) {
        Bm[c].h[0] = *(const v8h_t*)(myRow + c * 32 + off);
        Bm[c].h[1] = *(const v8h_t*)(myRow + c * 32 + 16 + off);
    }

    // Gather indices for the 3 tiles of 16: [pos 0..15], [neg 0..15], [neg 16..31]
    int idxRaw[3];
    idxRaw[0] = posIdx[(size_t)row * PP + m];
    idxRaw[1] = negIdx[(size_t)row * QQ + m];
    idxRaw[2] = negIdx[(size_t)row * QQ + 16 + m];

    #pragma unroll
    for (int t = 0; t < 3; ++t) {
        int r = idxRaw[t] < 0 ? 0 : idxRaw[t];           // clamp for gather
        const _Float16* grow = batchBase + (size_t)r * DD;
        v8f acc = {};
        #pragma unroll
        for (int c = 0; c < 4; ++c) {
            V16H A;
            A.h[0] = *(const v8h_t*)(grow + c * 32 + off);
            A.h[1] = *(const v8h_t*)(grow + c * 32 + 16 + off);
            acc = __builtin_amdgcn_wmma_f32_16x16x32_f16(
                      false, A.v, false, Bm[c].v, (short)0, acc, false, false);
        }
        // C/D layout: lanes 0-15 hold M=0..7 in acc[0..7] (col = lane),
        // lanes 16-31 hold M=8..15. All columns equal -> store unconditionally.
        #pragma unroll
        for (int i = 0; i < 8; ++i)
            scores[wave][t * 16 + off + i] = acc[i];
    }
    __syncthreads();

    // scores[0..15] = pos dots, scores[16..47] = neg dots.
    int   pI = idxRaw[0];                      // pos index for m = lane&15
    float s1 = scores[wave][lane];
    float s2 = (lane < 16) ? scores[wave][32 + lane] : 0.0f;

    // denominator: sum exp(pos masked to -inf -> 0) + sum exp(neg); T = 1
    float e1 = (lane < 16) ? ((pI >= 0) ? expf(s1) : 0.0f) : expf(s1);
    float e2 = (lane < 16) ? expf(s2) : 0.0f;
    float denom = e1 + e2;
    #pragma unroll
    for (int o = 16; o > 0; o >>= 1) denom += __shfl_xor(denom, o, 32);

    float lp = 0.0f, cnt = 0.0f;
    if (lane < 16 && pI >= 0) {
        lp  = -log1pf(e1 / denom);             // e1 == exp(pos_dist) here
        cnt = 1.0f;
    }
    float lpSum = lp, cntSum = cnt;
    #pragma unroll
    for (int o = 16; o > 0; o >>= 1) {
        lpSum  += __shfl_xor(lpSum,  o, 32);
        cntSum += __shfl_xor(cntSum, o, 32);
    }

    if (lane == 0) {
        float maskf   = (cntSum > 0.0f) ? 1.0f : 0.0f;
        float rowloss = (lpSum / cntSum) * maskf;   // 0/0 -> NaN like reference
        atomicAdd(&lossSum[b], rowloss);
        atomicAdd(&maskSum[b], maskf);
    }
}

// ---------------------------------------------------------------------------
// Kernel 3: scalar finalize: mean over B of lossSum[b]/maskSum[b].
// ---------------------------------------------------------------------------
__global__ void bcl_finalize_kernel(const float* __restrict__ lossSum,
                                    const float* __restrict__ maskSum,
                                    float* __restrict__ out) {
    int lane = threadIdx.x;
    float v = 0.0f;
    if (lane < BB) v = lossSum[lane] / maskSum[lane];
    #pragma unroll
    for (int o = 16; o > 0; o >>= 1) v += __shfl_xor(v, o, 32);
    if (lane == 0) out[0] = v / (float)BB;
}

// ---------------------------------------------------------------------------
extern "C" void kernel_launch(void* const* d_in, const int* in_sizes, int n_in,
                              void* d_out, int out_size, void* d_ws, size_t ws_size,
                              hipStream_t stream) {
    const float* feat = (const float*)d_in[0];
    const int*   pos  = (const int*)d_in[1];
    const int*   neg  = (const int*)d_in[2];

    // Workspace layout: [0, B*N*D*2) f16 normalized features; then 16 floats.
    _Float16* f16  = (_Float16*)d_ws;
    float*    sums = (float*)((char*)d_ws + (size_t)BB * NN * DD * sizeof(_Float16));
    float*    lossSum = sums;
    float*    maskSum = sums + BB;

    bcl_init_kernel<<<1, 64, 0, stream>>>(sums);
    bcl_normalize_kernel<<<(BB * NN) / 8, 256, 0, stream>>>(feat, f16);
    bcl_main_kernel<<<(BB * NN) / 8, 256, 0, stream>>>(f16, pos, neg, lossSum, maskSum);
    bcl_finalize_kernel<<<1, 32, 0, stream>>>(lossSum, maskSum, (float*)d_out);
}
